// TreeEncoder_56762287784427
// MI455X (gfx1250) — matrix operand order repaired and checked by reading, not compiled, not run on hardware
//
#include <hip/hip_runtime.h>
#include <hip/hip_bf16.h>

// ---------------------------------------------------------------------------
// Binary TreeLSTM over complete 16-level tree, fused per-level WMMA GEMM.
//   W_big[768,1280] bf16 (pre-packed in WMMA B-fragment layout "Bpack"):
//     rows   0..255 : x     -> [Wx_i, Wx_o, Wx_u, Wx_f, Wx_f]
//     rows 256..511 : h_left-> [Ui0 , Uo0 , Uu0 , Uf00, Uf10]
//     rows 512..767 : h_rght-> [Ui1 , Uo1 , Uu1 , Uf01, Uf11]
//   Per level:  P[m,1280] = A[m,768] @ W_big + bcat ; gates in registers.
//   Leaves: children are zero -> only K=0..255 and gate blocks i,o,u.
// ---------------------------------------------------------------------------

#define HH        256
#define N_NODES   65535
#define NTOT      1280         // 5 gate blocks * 256
#define NT_TILES  80           // 1280/16
#define KT_TILES  24           // 768/32

typedef __attribute__((ext_vector_type(16))) __bf16 v16bf;
typedef __attribute__((ext_vector_type(8)))  float  v8f;

union FragU {
    v16bf v;
    uint4 u[2];
};

__device__ __forceinline__ unsigned short f2bf(float f) {
    unsigned u = __float_as_uint(f);
    u += 0x7FFFu + ((u >> 16) & 1u);          // round-to-nearest-even
    return (unsigned short)(u >> 16);
}
__device__ __forceinline__ unsigned int pack2bf(float a, float b) {
    return (unsigned)f2bf(a) | ((unsigned)f2bf(b) << 16);
}
__device__ __forceinline__ float sigmoidf_(float x) {
    return 1.0f / (1.0f + __expf(-x));
}

// W_big[k][n] virtual fetch from the original parameter tensors.
__device__ __forceinline__ float wbig(int k, int n,
                                      const float* __restrict__ Wx,
                                      const float* __restrict__ Ui,
                                      const float* __restrict__ Uf,
                                      const float* __restrict__ Uo,
                                      const float* __restrict__ Uu) {
    const int rb = k >> 8, kr = k & 255;      // row block: 0=x,1=h_l,2=h_r
    const int cb = n >> 8, nc = n & 255;      // col block: i,o,u,f0,f1
    if (rb == 0) {
        // Wx column order is [i, f, o, u] (each 256 wide)
        int col = (cb == 0) ? nc
                : (cb == 1) ? 512 + nc
                : (cb == 2) ? 768 + nc
                            : 256 + nc;       // f block feeds both f0 and f1
        return Wx[kr * 1024 + col];
    }
    const int j = rb - 1;                     // child supplying h (0=left,1=right)
    if (cb == 0) return Ui[(j * 256 + kr) * 256 + nc];
    if (cb == 1) return Uo[(j * 256 + kr) * 256 + nc];
    if (cb == 2) return Uu[(j * 256 + kr) * 256 + nc];
    const int fj = cb - 3;                    // forget gate of child fj
    return Uf[((fj * 2 + j) * 256 + kr) * 256 + nc];
}

// Pack W_big into WMMA B-fragment layout:
// fragment (kt, nt): 32 lanes x 32B contiguous; lane L holds column n = nt*16+(L&15),
// K values kt*32 + (L>>4)*16 + {0..15}, packed 2 bf16 per dword.
__global__ void prep_weights(const float* __restrict__ Wx, const float* __restrict__ Ui,
                             const float* __restrict__ Uf, const float* __restrict__ Uo,
                             const float* __restrict__ Uu, unsigned int* __restrict__ Bpack) {
    int t = blockIdx.x * blockDim.x + threadIdx.x;
    if (t >= KT_TILES * NT_TILES * 32 * 8) return;
    const int v    = t & 7;
    const int L    = (t >> 3) & 31;
    const int rest = t >> 8;
    const int nt   = rest % NT_TILES;
    const int kt   = rest / NT_TILES;
    const int k    = kt * 32 + (L >> 4) * 16 + 2 * v;
    const int n    = nt * 16 + (L & 15);
    const float f0 = wbig(k,     n, Wx, Ui, Uf, Uo, Uu);
    const float f1 = wbig(k + 1, n, Wx, Ui, Uf, Uo, Uu);
    Bpack[(size_t)((kt * NT_TILES + nt) * 32 + L) * 8 + v] = pack2bf(f0, f1);
}

// bcat[1280] = [b_i, b_o, b_u, b_f, b_f]  (b layout is [i,f,o,u])
__global__ void prep_bias(const float* __restrict__ b, float* __restrict__ bcat) {
    int n = blockIdx.x * blockDim.x + threadIdx.x;
    if (n >= NTOT) return;
    const int cb = n >> 8, nc = n & 255;
    const int src = (cb == 0) ? nc
                  : (cb == 1) ? 512 + nc
                  : (cb == 2) ? 768 + nc
                              : 256 + nc;
    bcat[n] = b[src];
}

// xbf[node][0..255] = bf16(emb[tokens[node]])
__global__ void gather_x(const int* __restrict__ tokens, const float* __restrict__ emb,
                         unsigned int* __restrict__ xw) {
    int id = blockIdx.x * blockDim.x + threadIdx.x;     // node*128 + dword
    if (id >= N_NODES * 128) return;
    const int n  = id >> 7;
    const int cd = id & 127;
    const size_t base = (size_t)tokens[n] * 256 + (size_t)cd * 2;
    xw[(size_t)n * 128 + cd] = pack2bf(emb[base], emb[base + 1]);
}

__device__ __forceinline__ void loadB(FragU& f, const uint4* __restrict__ Bpack,
                                      int kt, int nt, int lane) {
    const uint4* bp = Bpack + ((size_t)(kt * NT_TILES + nt) * 32 + lane) * 2;
    f.u[0] = bp[0];
    f.u[1] = bp[1];
}

// One tree level: 32 nodes per block, 16 waves; wave w owns hidden cols [16w,16w+16)
// of each gate block so the gate math stays in registers.
template <bool LEAF>
__global__ __launch_bounds__(512, 1)
void tree_level(const uint4* __restrict__ Bpack, const float* __restrict__ bcat,
                const unsigned int* __restrict__ xw, float* __restrict__ hbuf,
                float* __restrict__ cbuf, float* __restrict__ rootH,
                float* __restrict__ rootC, int lo, int m) {
    constexpr int GMAX   = LEAF ? 3 : 5;   // leaves: forget gates unused (c_child = 0)
    constexpr int KSTEPS = LEAF ? 8 : KT_TILES;

    __shared__ __align__(16) unsigned short Atile[32][784];   // 768 + 16 pad

    const int tid       = threadIdx.x;
    const int node_base = lo + blockIdx.x * 32;
    const int mrem_raw  = m - blockIdx.x * 32;
    const int mrem      = mrem_raw < 32 ? mrem_raw : 32;

    // ---- stage x slab via async copy to LDS (16B per lane, ASYNCcnt-tracked) ----
    // 32 rows x 512B = 1024 chunks of 16B; 512 threads x 2.
    for (int i = 0; i < 2; ++i) {
        const int id  = i * 512 + tid;       // 0..1023
        const int row = id >> 5;             // 32 chunks per row
        const int c16 = id & 31;
        if (row < mrem) {
            const unsigned ldsoff =
                (unsigned)(unsigned long long)(uintptr_t)&Atile[row][c16 * 8];
            const unsigned long long gaddr =
                (unsigned long long)(uintptr_t)xw +
                ((unsigned long long)(node_base + row) * 512ull + (unsigned)(c16 * 16));
            asm volatile("global_load_async_to_lds_b128 %0, %1, off"
                         :: "v"(ldsoff), "v"(gaddr) : "memory");
        }
    }
    // zero-fill x rows of partial tiles (only top levels, single block)
    if (mrem < 32) {
        for (int id = tid; id < (32 - mrem) * 128; id += 512) {
            const int row = mrem + (id >> 7);
            const int cd  = id & 127;
            *((unsigned int*)&Atile[row][cd * 2]) = 0u;
        }
    }
    // ---- stage child hidden states, f32 -> bf16 (skip for leaves) ----
    if (!LEAF) {
        for (int i = 0; i < 16; ++i) {
            const int p     = i * 512 + tid;  // 0..8191 (pairs)
            const int row   = p >> 8;
            const int wi    = p & 255;
            const int child = wi >> 7;
            const int e     = (wi & 127) * 2;
            unsigned int val = 0u;
            if (row < mrem) {
                const int node = node_base + row;
                const float* hp = hbuf + (size_t)(2 * node + 1 + child) * HH + e;
                val = pack2bf(hp[0], hp[1]);
            }
            *((unsigned int*)&Atile[row][256 + child * 256 + e]) = val;
        }
    }
    asm volatile("s_wait_asynccnt 0x0" ::: "memory");
    __syncthreads();

    const int wv   = tid >> 5;                // wave 0..15 -> hidden col tile
    const int lane = tid & 31;
    const int lh   = lane >> 4;               // half-wave (K/M split)
    const int ll   = lane & 15;

    v8f acc[GMAX][2];
    const v8f vzero = {0.f, 0.f, 0.f, 0.f, 0.f, 0.f, 0.f, 0.f};
#pragma unroll
    for (int g = 0; g < GMAX; ++g)
        for (int mt = 0; mt < 2; ++mt) acc[g][mt] = vzero;

    // Software-pipelined B fragments: prefetch s+1 while WMMAs consume s.
    // The one-past-the-end prefetch reads inside the 2MB Bpack reservation (safe).
    FragU bnext;
    loadB(bnext, Bpack, 0, wv, lane);

    for (int kt = 0; kt < KSTEPS; ++kt) {
        // A fragments (contiguous K, matching Bpack): lane<16 K {0-15}, lane>=16 K {16-31}
        FragU a0, a1;
        {
            const unsigned short* r0 = &Atile[ll][kt * 32 + lh * 16];
            a0.u[0] = *(const uint4*)(r0);
            a0.u[1] = *(const uint4*)(r0 + 8);
            const unsigned short* r1 = &Atile[16 + ll][kt * 32 + lh * 16];
            a1.u[0] = *(const uint4*)(r1);
            a1.u[1] = *(const uint4*)(r1 + 8);
        }
#pragma unroll
        for (int g = 0; g < GMAX; ++g) {
            FragU bcur = bnext;                               // folds away under unroll
            const int ng  = (g == GMAX - 1) ? 0 : g + 1;
            const int nkt = (g == GMAX - 1) ? kt + 1 : kt;    // kt==KSTEPS overread is safe
            loadB(bnext, Bpack, nkt, ng * 16 + wv, lane);
            acc[g][0] = __builtin_amdgcn_wmma_f32_16x16x32_bf16(
                false, a0.v, false, bcur.v, (short)0, acc[g][0], false, false);
            acc[g][1] = __builtin_amdgcn_wmma_f32_16x16x32_bf16(
                false, a1.v, false, bcur.v, (short)0, acc[g][1], false, false);
        }
    }

    // ---- gate math, fully in registers ----
    const int hid = wv * 16 + ll;
    const float bi  = bcat[hid];
    const float bo  = bcat[256 + hid];
    const float bu  = bcat[512 + hid];
    const float bfg = bcat[768 + hid];

#pragma unroll
    for (int mt = 0; mt < 2; ++mt) {
#pragma unroll
        for (int v = 0; v < 8; ++v) {
            const int row = mt * 16 + lh * 8 + v;     // C/D layout: lane half -> M+8
            if (row >= mrem) continue;
            const int node = node_base + row;
            const float gi = sigmoidf_(acc[0][mt][v] + bi);
            const float go = sigmoidf_(acc[1][mt][v] + bo);
            const float gu = tanhf(acc[2][mt][v] + bu);
            float cn;
            if (!LEAF) {
                const float cl = cbuf[(size_t)(2 * node + 1) * HH + hid];
                const float cr = cbuf[(size_t)(2 * node + 2) * HH + hid];
                const float f0 = sigmoidf_(acc[GMAX - 2][mt][v] + bfg);
                const float f1 = sigmoidf_(acc[GMAX - 1][mt][v] + bfg);
                cn = gi * gu + f0 * cl + f1 * cr;
            } else {
                cn = gi * gu;
            }
            const float hn = go * tanhf(cn);
            hbuf[(size_t)node * HH + hid] = hn;
            cbuf[(size_t)node * HH + hid] = cn;
            if (node == 0) { rootH[hid] = hn; rootC[hid] = cn; }
        }
    }
}

extern "C" void kernel_launch(void* const* d_in, const int* in_sizes, int n_in,
                              void* d_out, int out_size, void* d_ws, size_t ws_size,
                              hipStream_t stream) {
    (void)in_sizes; (void)n_in; (void)out_size; (void)ws_size;
    const int*   tokens = (const int*)d_in[0];
    const float* emb    = (const float*)d_in[1];
    const float* Wx     = (const float*)d_in[2];
    const float* b      = (const float*)d_in[3];
    const float* Ui     = (const float*)d_in[4];
    const float* Uf     = (const float*)d_in[5];
    const float* Uo     = (const float*)d_in[6];
    const float* Uu     = (const float*)d_in[7];

    char* ws = (char*)d_ws;
    unsigned int* Bpack = (unsigned int*)ws;                               // 1.97 MB (2 MB resv)
    float*        bcat  = (float*)(ws + (size_t)2 * 1024 * 1024);          // 5 KB
    unsigned int* xw    = (unsigned int*)(ws + (size_t)2 * 1024 * 1024 + 64 * 1024);   // 32 MB
    float*        cbuf  = (float*)(ws + (size_t)2 * 1024 * 1024 + 64 * 1024
                                      + (size_t)33554432);                 // 64 MB

    float* hbuf  = (float*)d_out;                  // annotations, heap order
    float* rootH = hbuf + (size_t)N_NODES * HH;    // hiddens[0]
    float* rootC = rootH + HH;                     // cell_states[0]

    prep_weights<<<dim3(1920), dim3(256), 0, stream>>>(Wx, Ui, Uf, Uo, Uu, Bpack);
    prep_bias<<<dim3(5), dim3(256), 0, stream>>>(b, bcat);
    gather_x<<<dim3(32768), dim3(256), 0, stream>>>(tokens, emb, xw);

    // Leaf level (d=15): K=256, gates i,o,u only.
    {
        const int lo = (1 << 15) - 1, m = 1 << 15;
        tree_level<true><<<dim3((m + 31) / 32), dim3(512), 0, stream>>>(
            (const uint4*)Bpack, bcat, xw, hbuf, cbuf, rootH, rootC, lo, m);
    }
    // Internal levels (d=14..0): K=768, all 5 gate blocks.
    for (int d = 14; d >= 0; --d) {
        const int lo = (1 << d) - 1;
        const int m  = 1 << d;
        tree_level<false><<<dim3((m + 31) / 32), dim3(512), 0, stream>>>(
            (const uint4*)Bpack, bcat, xw, hbuf, cbuf, rootH, rootC, lo, m);
    }
}
